// TOEncoder_29626684408323
// MI455X (gfx1250) — matrix operand order repaired and checked by reading, not compiled, not run on hardware
//
#include <hip/hip_runtime.h>
#include <hip/hip_bf16.h>
#include <stdint.h>

// Problem constants (fixed by setup_inputs)
#define B_   8
#define NQ   1024
#define MM   1024
#define CC   640
#define EPS  0.05f
#define INV_EPS 20.0f
#define NEGC (-1.0e9f)

typedef __attribute__((ext_vector_type(16))) __bf16 bf16x16;
typedef __attribute__((ext_vector_type(8)))  __bf16 bf16x8;
typedef __attribute__((ext_vector_type(2)))  __bf16 bf16x2;
typedef __attribute__((ext_vector_type(8)))  float  f32x8;

// ---------- conversions (native bf16 path on gfx1250) ----------
__device__ __forceinline__ __bf16 f2bf(float f) { return (__bf16)f; }
__device__ __forceinline__ float bf2f(__bf16 h) {
    unsigned short s = __builtin_bit_cast(unsigned short, h);
    unsigned u = ((unsigned)s) << 16;
    return __builtin_bit_cast(float, u);
}
__device__ __forceinline__ unsigned packbf2(float x, float y) {
#if __has_builtin(__builtin_amdgcn_cvt_pk_bf16_f32)
    bf16x2 p = __builtin_amdgcn_cvt_pk_bf16_f32(x, y);
    return __builtin_bit_cast(unsigned, p);
#else
    unsigned lo = (unsigned)__builtin_bit_cast(unsigned short, f2bf(x));
    unsigned hi = (unsigned)__builtin_bit_cast(unsigned short, f2bf(y));
    return lo | (hi << 16);
#endif
}
__device__ __forceinline__ void st_bf4(__bf16* d, float4 f) {   // 4 f32 -> 4 bf16 (b64 store)
    uint2 p; p.x = packbf2(f.x, f.y); p.y = packbf2(f.z, f.w);
    *(uint2*)d = p;
}
__device__ __forceinline__ __bf16 bfbits(unsigned v) {
    return __builtin_bit_cast(__bf16, (unsigned short)(v & 0xFFFFu));
}

// ---------- CDNA5 async global->LDS copy (ASYNCcnt path) ----------
__device__ __forceinline__ unsigned lds_addr(const void* p) {
    return (unsigned)(uintptr_t)(__attribute__((address_space(3))) const char*)p;
}
__device__ __forceinline__ void async_cp16(unsigned lds_byte, const void* g) {
    asm volatile("global_load_async_to_lds_b128 %0, %1, off"
                 :: "v"(lds_byte), "v"((unsigned long long)(uintptr_t)g)
                 : "memory");
}
__device__ __forceinline__ void wait_async0() {
    asm volatile("s_wait_asynccnt 0x0" ::: "memory");
}

// LDS tile pitch in bf16 elements: 32 K + 8 pad (80B rows -> conflict-free 16B frag reads)
#define LDP 40

// A fragment (16x32 bf16): lane -> row M=l&15, K = g*16 .. g*16+15
__device__ __forceinline__ bf16x16 frag_a(const __bf16* tile, int lane) {
    int row = lane & 15, g = lane >> 4;
    const __bf16* base = tile + row * LDP;
    bf16x8 lo = *(const bf16x8*)(base + g * 16);
    bf16x8 hi = *(const bf16x8*)(base + g * 16 + 8);
    return __builtin_shufflevector(lo, hi, 0,1,2,3,4,5,6,7,8,9,10,11,12,13,14,15);
}
// B fragment (32x16 bf16): lane -> col N=l&15, K = g*16 .. g*16+15 (2 per VGPR)
__device__ __forceinline__ bf16x16 frag_b(const __bf16* tile, int lane) {
    int col = lane & 15, g = lane >> 4;
    const __bf16* base = tile + col * LDP + g * 16;
    bf16x8 lo = *(const bf16x8*)(base);
    bf16x8 hi = *(const bf16x8*)(base + 8);
    return __builtin_shufflevector(lo, hi, 0,1,2,3,4,5,6,7,8,9,10,11,12,13,14,15);
}
__device__ __forceinline__ f32x8 wmma_bf16(bf16x16 a, bf16x16 b, f32x8 c) {
    return __builtin_amdgcn_wmma_f32_16x16x32_bf16(false, a, false, b, (short)0, c, false, false);
}
__device__ __forceinline__ void wmma_tile(const __bf16* As, const __bf16* Bs,
                                          int wm, int wn, int lane, f32x8 acc[2][2]) {
    bf16x16 a0 = frag_a(As + (wm * 32 + 0) * LDP, lane);
    bf16x16 a1 = frag_a(As + (wm * 32 + 16) * LDP, lane);
    bf16x16 b0 = frag_b(Bs + (wn * 32 + 0) * LDP, lane);
    bf16x16 b1 = frag_b(Bs + (wn * 32 + 16) * LDP, lane);
    acc[0][0] = wmma_bf16(a0, b0, acc[0][0]);
    acc[0][1] = wmma_bf16(a0, b1, acc[0][1]);
    acc[1][0] = wmma_bf16(a1, b0, acc[1][0]);
    acc[1][1] = wmma_bf16(a1, b1, acc[1][1]);
}

// online logsumexp accumulate/merge
__device__ __forceinline__ void lse_acc(float K, float& mx, float& sm) {
    float nm = fmaxf(mx, K);
    sm = sm * __expf(mx - nm) + __expf(K - nm);
    mx = nm;
}
__device__ __forceinline__ void lse_merge(float mo, float so, float& mx, float& sm) {
    float nm = fmaxf(mx, mo);
    sm = sm * __expf(mx - nm) + so * __expf(mo - nm);
    mx = nm;
}

// =====================================================================
// Setup: counts[b], log_mu[b,m], u=v=0
// =====================================================================
__global__ __launch_bounds__(256) void k_setup_counts(const unsigned char* __restrict__ mask,
                                                      float* __restrict__ counts) {
    int w = threadIdx.x >> 5, lane = threadIdx.x & 31;   // wave w handles batch w
    int c = 0;
    for (int i = 0; i < 32; ++i) c += mask[w * MM + lane + 32 * i] ? 1 : 0;
    #pragma unroll
    for (int off = 16; off; off >>= 1) c += __shfl_xor(c, off);
    if (lane == 0) counts[w] = (float)c;
}

__global__ __launch_bounds__(256) void k_setup_vecs(const unsigned char* __restrict__ mask,
                                                    const float* __restrict__ counts,
                                                    float* __restrict__ log_mu,
                                                    float* __restrict__ u, float* __restrict__ v) {
    int i = blockIdx.x * 256 + threadIdx.x;              // 0..8191
    int b = i >> 10;
    float cnt = counts[b];
    float xx = mask[i] ? (1.0f / fmaxf(cnt, 1.0f)) : 0.0f;
    if (cnt == 0.0f) xx = 1.0f / (float)MM;
    log_mu[i] = __logf(xx + 1e-8f);
    u[i] = 0.0f;
    v[i] = 0.0f;
}

// =====================================================================
// Fused linear: Y(bf16) = X(f32, 8192 x 640) @ W^T + bias
// block tile 64(M) x 128(N), 8 waves (2x4), wave tile 32x32, K-step 32
// Double-buffered LDS + software-pipelined register staging.
// =====================================================================
__global__ __launch_bounds__(256) void k_linear_bf16(const float* __restrict__ X,
                                                     const float* __restrict__ W,
                                                     const float* __restrict__ bias,
                                                     __bf16* __restrict__ Y) {
    __shared__ __align__(16) __bf16 As[2][64 * LDP];
    __shared__ __align__(16) __bf16 Bs[2][128 * LDP];
    const int tid = threadIdx.x, lane = tid & 31, w = tid >> 5;
    const int wm = w & 1, wn = w >> 1;
    const int m0 = blockIdx.x * 64, n0 = blockIdx.y * 128;
    const int rA = tid >> 3, cA = (tid & 7) << 2;        // 32 rows/pass, float4 chunks
    const float* Ap = X + (size_t)(m0 + rA) * CC + cA;
    const float* Bp = W + (size_t)(n0 + rA) * CC + cA;
    f32x8 acc[2][2] = {};
    float4 a0 = *(const float4*)(Ap);
    float4 a1 = *(const float4*)(Ap + (size_t)32 * CC);
    float4 b0 = *(const float4*)(Bp);
    float4 b1 = *(const float4*)(Bp + (size_t)32 * CC);
    float4 b2 = *(const float4*)(Bp + (size_t)64 * CC);
    float4 b3 = *(const float4*)(Bp + (size_t)96 * CC);
    st_bf4(&As[0][rA * LDP + cA], a0);
    st_bf4(&As[0][(rA + 32) * LDP + cA], a1);
    st_bf4(&Bs[0][rA * LDP + cA], b0);
    st_bf4(&Bs[0][(rA + 32) * LDP + cA], b1);
    st_bf4(&Bs[0][(rA + 64) * LDP + cA], b2);
    st_bf4(&Bs[0][(rA + 96) * LDP + cA], b3);
    __syncthreads();
    int buf = 0;
    for (int k0 = 32; k0 < CC; k0 += 32) {
        a0 = *(const float4*)(Ap + k0);                  // issue next-tile loads
        a1 = *(const float4*)(Ap + (size_t)32 * CC + k0);
        b0 = *(const float4*)(Bp + k0);
        b1 = *(const float4*)(Bp + (size_t)32 * CC + k0);
        b2 = *(const float4*)(Bp + (size_t)64 * CC + k0);
        b3 = *(const float4*)(Bp + (size_t)96 * CC + k0);
        wmma_tile(As[buf], Bs[buf], wm, wn, lane, acc);  // math hides load latency
        int nb = buf ^ 1;
        st_bf4(&As[nb][rA * LDP + cA], a0);
        st_bf4(&As[nb][(rA + 32) * LDP + cA], a1);
        st_bf4(&Bs[nb][rA * LDP + cA], b0);
        st_bf4(&Bs[nb][(rA + 32) * LDP + cA], b1);
        st_bf4(&Bs[nb][(rA + 64) * LDP + cA], b2);
        st_bf4(&Bs[nb][(rA + 96) * LDP + cA], b3);
        __syncthreads();
        buf = nb;
    }
    wmma_tile(As[buf], Bs[buf], wm, wn, lane, acc);
    const int g = lane >> 4, nl = lane & 15;
    #pragma unroll
    for (int ti = 0; ti < 2; ++ti)
        #pragma unroll
        for (int tj = 0; tj < 2; ++tj) {
            int col = n0 + wn * 32 + tj * 16 + nl;
            float bb = bias[col];
            #pragma unroll
            for (int r = 0; r < 8; ++r) {
                int row = m0 + wm * 32 + ti * 16 + g * 8 + r;
                Y[(size_t)row * CC + col] = f2bf(acc[ti][tj][r] + bb);
            }
        }
}

// =====================================================================
// Row rsqrt-norms over bf16 rows (wave per row); folded into sim epilogue
// =====================================================================
__global__ __launch_bounds__(256) void k_rownorm(const __bf16* __restrict__ Y,
                                                 float* __restrict__ rn) {
    int w = threadIdx.x >> 5, lane = threadIdx.x & 31;
    int row = blockIdx.x * 8 + w;
    const __bf16* p = Y + (size_t)row * CC;
    float s = 0.0f;
    for (int i = lane; i < CC; i += 32) { float f = bf2f(p[i]); s += f * f; }
    #pragma unroll
    for (int off = 16; off; off >>= 1) s += __shfl_xor(s, off);
    if (lane == 0) rn[row] = 1.0f / fmaxf(sqrtf(s), 1e-12f);
}

// =====================================================================
// sim[b,m,n] = rnk[b,m]*rnq[n,b]*(k_bm . q_nb)   also writes simT[b,n,m]
// Pure bf16 tile copies -> async global->LDS (ASYNCcnt), double-buffered.
// grid (16, 8, 8): m-tile(64), n-tile(128), batch
// =====================================================================
__global__ __launch_bounds__(256) void k_sim(const __bf16* __restrict__ Kb,
                                             const __bf16* __restrict__ Qb,
                                             const float* __restrict__ rnk,
                                             const float* __restrict__ rnq,
                                             float* __restrict__ sim,
                                             float* __restrict__ simT) {
    __shared__ __align__(16) __bf16 As[2][64 * LDP];
    __shared__ __align__(16) __bf16 Bs[2][128 * LDP];
    const int tid = threadIdx.x, lane = tid & 31, w = tid >> 5;
    const int wm = w & 1, wn = w >> 1;
    const int b = blockIdx.z;
    const int m0 = blockIdx.x * 64, n0 = blockIdx.y * 128;
    const __bf16* Kbase = Kb + (size_t)b * MM * CC;
    const int rA = tid >> 2, c8 = (tid & 3) << 3;        // 16B chunk per thread
    const __bf16* gA  = Kbase + (size_t)(m0 + rA) * CC + c8;
    const __bf16* gB0 = Qb + ((size_t)(n0 + rA) * B_ + b) * CC + c8;
    const __bf16* gB1 = Qb + ((size_t)(n0 + rA + 64) * B_ + b) * CC + c8;
    unsigned ldsA[2]  = { lds_addr(&As[0][rA * LDP + c8]),        lds_addr(&As[1][rA * LDP + c8]) };
    unsigned ldsB0[2] = { lds_addr(&Bs[0][rA * LDP + c8]),        lds_addr(&Bs[1][rA * LDP + c8]) };
    unsigned ldsB1[2] = { lds_addr(&Bs[0][(rA + 64) * LDP + c8]), lds_addr(&Bs[1][(rA + 64) * LDP + c8]) };
    f32x8 acc[2][2] = {};
    async_cp16(ldsA[0], gA);
    async_cp16(ldsB0[0], gB0);
    async_cp16(ldsB1[0], gB1);
    wait_async0();
    __syncthreads();
    int buf = 0;
    for (int k0 = 32; k0 < CC; k0 += 32) {
        int nb = buf ^ 1;
        async_cp16(ldsA[nb], gA + k0);                   // DMA next tile, overlapped
        async_cp16(ldsB0[nb], gB0 + k0);
        async_cp16(ldsB1[nb], gB1 + k0);
        wmma_tile(As[buf], Bs[buf], wm, wn, lane, acc);
        wait_async0();
        __syncthreads();
        buf = nb;
    }
    wmma_tile(As[buf], Bs[buf], wm, wn, lane, acc);
    const int g = lane >> 4, nl = lane & 15;
    #pragma unroll
    for (int ti = 0; ti < 2; ++ti)
        #pragma unroll
        for (int tj = 0; tj < 2; ++tj) {
            int coln = n0 + wn * 32 + tj * 16 + nl;
            float sq = rnq[(size_t)coln * B_ + b];
            #pragma unroll
            for (int r = 0; r < 8; ++r) {
                int rowm = m0 + wm * 32 + ti * 16 + g * 8 + r;
                float val = acc[ti][tj][r] * rnk[b * MM + rowm] * sq;
                sim [((size_t)b * MM + rowm) * NQ + coln] = val;
                simT[((size_t)b * NQ + coln) * MM + rowm] = val;
            }
        }
}

// =====================================================================
// Sinkhorn half-iterations: wave-per-row online logsumexp, L2-resident rows
// =====================================================================
__global__ __launch_bounds__(256) void k_u_update(const float* __restrict__ sim,
                                                  const unsigned char* __restrict__ mask,
                                                  float* __restrict__ u,
                                                  const float* __restrict__ v,
                                                  const float* __restrict__ log_mu) {
    int w = threadIdx.x >> 5, lane = threadIdx.x & 31;
    int r = blockIdx.x * 8 + w;                          // (b,m) flat
    int b = r >> 10;
    bool mr = mask[r] != 0;
    float u_r = u[r];
    const float* srow = sim + (size_t)r * NQ;
    const float* vv = v + b * NQ;
    float c0 = u_r - 1.0f;
    float mx = -3.0e38f, sm = 0.0f;
    for (int i = 0; i < 8; ++i) {
        int base = i * 128 + lane * 4;
        float4 s4 = *(const float4*)(srow + base);
        float4 v4 = *(const float4*)(vv + base);
        lse_acc(mr ? (s4.x + v4.x + c0) * INV_EPS : NEGC, mx, sm);
        lse_acc(mr ? (s4.y + v4.y + c0) * INV_EPS : NEGC, mx, sm);
        lse_acc(mr ? (s4.z + v4.z + c0) * INV_EPS : NEGC, mx, sm);
        lse_acc(mr ? (s4.w + v4.w + c0) * INV_EPS : NEGC, mx, sm);
    }
    #pragma unroll
    for (int off = 16; off; off >>= 1)
        lse_merge(__shfl_xor(mx, off), __shfl_xor(sm, off), mx, sm);
    if (lane == 0) {
        float lse = mx + __logf(sm);
        u[r] = EPS * (log_mu[r] - lse) + u_r;
    }
}

__global__ __launch_bounds__(256) void k_v_update(const float* __restrict__ simT,
                                                  const unsigned char* __restrict__ mask,
                                                  const float* __restrict__ u,
                                                  float* __restrict__ v) {
    int w = threadIdx.x >> 5, lane = threadIdx.x & 31;
    int r = blockIdx.x * 8 + w;                          // (b,n) flat
    int b = r >> 10;
    float v_c = v[r];
    const float* srow = simT + (size_t)r * MM;
    const float* ub = u + b * MM;
    const unsigned char* mb = mask + b * MM;
    float c0 = v_c - 1.0f;
    float mx = -3.0e38f, sm = 0.0f;
    for (int i = 0; i < 8; ++i) {
        int base = i * 128 + lane * 4;
        float4 s4 = *(const float4*)(srow + base);
        float4 u4 = *(const float4*)(ub + base);
        uchar4 m4 = *(const uchar4*)(mb + base);
        lse_acc(m4.x ? (s4.x + u4.x + c0) * INV_EPS : NEGC, mx, sm);
        lse_acc(m4.y ? (s4.y + u4.y + c0) * INV_EPS : NEGC, mx, sm);
        lse_acc(m4.z ? (s4.z + u4.z + c0) * INV_EPS : NEGC, mx, sm);
        lse_acc(m4.w ? (s4.w + u4.w + c0) * INV_EPS : NEGC, mx, sm);
    }
    #pragma unroll
    for (int off = 16; off; off >>= 1)
        lse_merge(__shfl_xor(mx, off), __shfl_xor(sm, off), mx, sm);
    if (lane == 0) {
        float log_nu = __logf(1.0f / (float)NQ + 1e-8f);
        float lse = mx + __logf(sm);
        v[r] = EPS * (log_nu - lse) + v_c;
    }
}

// =====================================================================
// x[n,b,:] = sum_m T[b,m,n] * v[b,m,:]  (T rebuilt on the fly from simT)
// grid (16, 5, 8): n-tile(64), c-tile(128), batch. Double-buffered/pipelined.
// =====================================================================
__device__ __forceinline__ void st_T4(__bf16* dst, float4 s, const float* um,
                                      const unsigned char* mm4, float vn) {
    float t0 = mm4[0] ? __expf((s.x + um[0] + vn - 1.0f) * INV_EPS) : 0.0f;
    float t1 = mm4[1] ? __expf((s.y + um[1] + vn - 1.0f) * INV_EPS) : 0.0f;
    float t2 = mm4[2] ? __expf((s.z + um[2] + vn - 1.0f) * INV_EPS) : 0.0f;
    float t3 = mm4[3] ? __expf((s.w + um[3] + vn - 1.0f) * INV_EPS) : 0.0f;
    uint2 p; p.x = packbf2(t0, t1); p.y = packbf2(t2, t3);
    *(uint2*)dst = p;
}
__device__ __forceinline__ void st_vT(__bf16* BsBuf, int mm, int c8, uint4 d) {
    BsBuf[(c8 + 0) * LDP + mm] = bfbits(d.x);
    BsBuf[(c8 + 1) * LDP + mm] = bfbits(d.x >> 16);
    BsBuf[(c8 + 2) * LDP + mm] = bfbits(d.y);
    BsBuf[(c8 + 3) * LDP + mm] = bfbits(d.y >> 16);
    BsBuf[(c8 + 4) * LDP + mm] = bfbits(d.z);
    BsBuf[(c8 + 5) * LDP + mm] = bfbits(d.z >> 16);
    BsBuf[(c8 + 6) * LDP + mm] = bfbits(d.w);
    BsBuf[(c8 + 7) * LDP + mm] = bfbits(d.w >> 16);
}

__global__ __launch_bounds__(256) void k_tv(const float* __restrict__ simT,
                                            const __bf16* __restrict__ Vb,
                                            const unsigned char* __restrict__ mask,
                                            const float* __restrict__ u,
                                            const float* __restrict__ vvec,
                                            __bf16* __restrict__ Xb) {
    __shared__ __align__(16) __bf16 As[2][64 * LDP];
    __shared__ __align__(16) __bf16 Bs[2][128 * LDP];
    const int tid = threadIdx.x, lane = tid & 31, w = tid >> 5;
    const int wm = w & 1, wn = w >> 1;
    const int b = blockIdx.z;
    const int n0 = blockIdx.x * 64, c0 = blockIdx.y * 128;
    const float* sT = simT + (size_t)b * NQ * MM;
    const __bf16* Vbb = Vb + (size_t)b * MM * CC;
    const float* ub = u + b * MM;
    const float* vb = vvec + b * NQ;
    const unsigned char* mb = mask + b * MM;
    // A: 64(n) x 32(m) from simT, thread -> rows nnA, nnA+32, float4 along m
    const int nnA = tid >> 3, mcA = (tid & 7) << 2;
    const float* sP = sT + (size_t)(n0 + nnA) * MM + mcA;
    const float vn0 = vb[n0 + nnA];
    const float vn1 = vb[n0 + nnA + 32];
    // B: 32(m) x 128(c) from vbf, transposed into Bs[c][m]
    const int mmB = tid >> 4, c8B = (tid & 15) << 3;
    const __bf16* vP = Vbb + (size_t)mmB * CC + c0 + c8B;
    f32x8 acc[2][2] = {};
    float4 s0 = *(const float4*)(sP);
    float4 s1 = *(const float4*)(sP + (size_t)32 * MM);
    uint4 w0 = *(const uint4*)(vP);
    uint4 w1 = *(const uint4*)(vP + (size_t)16 * CC);
    st_T4(&As[0][nnA * LDP + mcA],        s0, ub + mcA, mb + mcA, vn0);
    st_T4(&As[0][(nnA + 32) * LDP + mcA], s1, ub + mcA, mb + mcA, vn1);
    st_vT(Bs[0], mmB, c8B, w0);
    st_vT(Bs[0], mmB + 16, c8B, w1);
    __syncthreads();
    int buf = 0;
    for (int m0 = 32; m0 < MM; m0 += 32) {
        s0 = *(const float4*)(sP + m0);
        s1 = *(const float4*)(sP + (size_t)32 * MM + m0);
        w0 = *(const uint4*)(vP + (size_t)m0 * CC);
        w1 = *(const uint4*)(vP + (size_t)(m0 + 16) * CC);
        wmma_tile(As[buf], Bs[buf], wm, wn, lane, acc);
        int nb = buf ^ 1;
        st_T4(&As[nb][nnA * LDP + mcA],        s0, ub + m0 + mcA, mb + m0 + mcA, vn0);
        st_T4(&As[nb][(nnA + 32) * LDP + mcA], s1, ub + m0 + mcA, mb + m0 + mcA, vn1);
        st_vT(Bs[nb], mmB, c8B, w0);
        st_vT(Bs[nb], mmB + 16, c8B, w1);
        __syncthreads();
        buf = nb;
    }
    wmma_tile(As[buf], Bs[buf], wm, wn, lane, acc);
    const int g = lane >> 4, nl = lane & 15;
    #pragma unroll
    for (int ti = 0; ti < 2; ++ti)
        #pragma unroll
        for (int tj = 0; tj < 2; ++tj) {
            int col = c0 + wn * 32 + tj * 16 + nl;
            #pragma unroll
            for (int r = 0; r < 8; ++r) {
                int rown = n0 + wm * 32 + ti * 16 + g * 8 + r;
                Xb[((size_t)rown * B_ + b) * CC + col] = f2bf(acc[ti][tj][r]);
            }
        }
}

// =====================================================================
// out(f32) = x(bf16) @ Wp^T + bp   (A tile via async copy, B via regs)
// =====================================================================
__global__ __launch_bounds__(256) void k_proj_f32(const __bf16* __restrict__ Xb,
                                                  const float* __restrict__ W,
                                                  const float* __restrict__ bias,
                                                  float* __restrict__ Y) {
    __shared__ __align__(16) __bf16 As[2][64 * LDP];
    __shared__ __align__(16) __bf16 Bs[2][128 * LDP];
    const int tid = threadIdx.x, lane = tid & 31, w = tid >> 5;
    const int wm = w & 1, wn = w >> 1;
    const int m0 = blockIdx.x * 64, n0 = blockIdx.y * 128;
    const int rA = tid >> 2, c8 = (tid & 3) << 3;        // A: bf16 16B chunks (async)
    const int rB = tid >> 3, cB = (tid & 7) << 2;        // B: f32 float4 chunks
    const __bf16* gA = Xb + (size_t)(m0 + rA) * CC + c8;
    const float* Bp = W + (size_t)(n0 + rB) * CC + cB;
    unsigned ldsA[2] = { lds_addr(&As[0][rA * LDP + c8]), lds_addr(&As[1][rA * LDP + c8]) };
    f32x8 acc[2][2] = {};
    async_cp16(ldsA[0], gA);
    float4 b0 = *(const float4*)(Bp);
    float4 b1 = *(const float4*)(Bp + (size_t)32 * CC);
    float4 b2 = *(const float4*)(Bp + (size_t)64 * CC);
    float4 b3 = *(const float4*)(Bp + (size_t)96 * CC);
    st_bf4(&Bs[0][rB * LDP + cB], b0);
    st_bf4(&Bs[0][(rB + 32) * LDP + cB], b1);
    st_bf4(&Bs[0][(rB + 64) * LDP + cB], b2);
    st_bf4(&Bs[0][(rB + 96) * LDP + cB], b3);
    wait_async0();
    __syncthreads();
    int buf = 0;
    for (int k0 = 32; k0 < CC; k0 += 32) {
        int nb = buf ^ 1;
        async_cp16(ldsA[nb], gA + k0);
        b0 = *(const float4*)(Bp + k0);
        b1 = *(const float4*)(Bp + (size_t)32 * CC + k0);
        b2 = *(const float4*)(Bp + (size_t)64 * CC + k0);
        b3 = *(const float4*)(Bp + (size_t)96 * CC + k0);
        wmma_tile(As[buf], Bs[buf], wm, wn, lane, acc);
        st_bf4(&Bs[nb][rB * LDP + cB], b0);
        st_bf4(&Bs[nb][(rB + 32) * LDP + cB], b1);
        st_bf4(&Bs[nb][(rB + 64) * LDP + cB], b2);
        st_bf4(&Bs[nb][(rB + 96) * LDP + cB], b3);
        wait_async0();
        __syncthreads();
        buf = nb;
    }
    wmma_tile(As[buf], Bs[buf], wm, wn, lane, acc);
    const int g = lane >> 4, nl = lane & 15;
    #pragma unroll
    for (int ti = 0; ti < 2; ++ti)
        #pragma unroll
        for (int tj = 0; tj < 2; ++tj) {
            int col = n0 + wn * 32 + tj * 16 + nl;
            float bb = bias[col];
            #pragma unroll
            for (int r = 0; r < 8; ++r) {
                int row = m0 + wm * 32 + ti * 16 + g * 8 + r;
                Y[(size_t)row * CC + col] = acc[ti][tj][r] + bb;
            }
        }
}

// =====================================================================
// attn_save[b,m] = mask ? M*Nq * sum_n sim*T : 0   (wave per row)
// =====================================================================
__global__ __launch_bounds__(256) void k_attn(const float* __restrict__ sim,
                                              const unsigned char* __restrict__ mask,
                                              const float* __restrict__ u,
                                              const float* __restrict__ v,
                                              float* __restrict__ attn) {
    int w = threadIdx.x >> 5, lane = threadIdx.x & 31;
    int r = blockIdx.x * 8 + w;                          // (b,m) flat
    int b = r >> 10;
    bool mr = mask[r] != 0;
    if (!mr) { if (lane == 0) attn[r] = 0.0f; return; }  // wave-uniform branch
    float u_r = u[r];
    const float* srow = sim + (size_t)r * NQ;
    const float* vv = v + b * NQ;
    float c0 = u_r - 1.0f;
    float acc = 0.0f;
    for (int i = 0; i < 8; ++i) {
        int base = i * 128 + lane * 4;
        float4 s4 = *(const float4*)(srow + base);
        float4 v4 = *(const float4*)(vv + base);
        acc += s4.x * __expf((s4.x + v4.x + c0) * INV_EPS);
        acc += s4.y * __expf((s4.y + v4.y + c0) * INV_EPS);
        acc += s4.z * __expf((s4.z + v4.z + c0) * INV_EPS);
        acc += s4.w * __expf((s4.w + v4.w + c0) * INV_EPS);
    }
    #pragma unroll
    for (int off = 16; off; off >>= 1) acc += __shfl_xor(acc, off);
    if (lane == 0) attn[r] = acc * (float)((size_t)MM * NQ);
}

// =====================================================================
// Host orchestration
// =====================================================================
extern "C" void kernel_launch(void* const* d_in, const int* in_sizes, int n_in,
                              void* d_out, int out_size, void* d_ws, size_t ws_size,
                              hipStream_t stream) {
    (void)in_sizes; (void)n_in; (void)out_size; (void)ws_size;
    const float* xq = (const float*)d_in[0];
    const float* xk = (const float*)d_in[1];
    const float* xv = (const float*)d_in[2];
    const unsigned char* mask = (const unsigned char*)d_in[3];  // jax bool = 1 byte
    const float* Wq = (const float*)d_in[4];
    const float* bq = (const float*)d_in[5];
    const float* Wk = (const float*)d_in[6];
    const float* bk = (const float*)d_in[7];
    const float* Wv = (const float*)d_in[8];
    const float* bv = (const float*)d_in[9];
    const float* Wp = (const float*)d_in[10];
    const float* bp = (const float*)d_in[11];

    float* out  = (float*)d_out;                          // (Nq,B,C) flat
    float* attn = out + (size_t)NQ * B_ * CC;             // (B,M) flat

    // workspace arena (256B aligned slices)
    char* ws = (char*)d_ws;
    auto arena = [&](size_t bytes) -> void* {
        void* p = (void*)ws;
        ws += (bytes + 255) & ~(size_t)255;
        return p;
    };
    const size_t R = 8192;                                // 8192 rows everywhere
    __bf16* qbf    = (__bf16*)arena(R * CC * sizeof(__bf16));
    __bf16* kbf    = (__bf16*)arena(R * CC * sizeof(__bf16));
    __bf16* vbf    = (__bf16*)arena(R * CC * sizeof(__bf16));
    __bf16* xbf    = (__bf16*)arena(R * CC * sizeof(__bf16));
    float*  sim    = (float*)arena((size_t)B_ * MM * NQ * sizeof(float));
    float*  simT   = (float*)arena((size_t)B_ * NQ * MM * sizeof(float));
    float*  u      = (float*)arena(R * sizeof(float));
    float*  v      = (float*)arena(R * sizeof(float));
    float*  log_mu = (float*)arena(R * sizeof(float));
    float*  counts = (float*)arena(B_ * sizeof(float));
    float*  rnq    = (float*)arena(R * sizeof(float));
    float*  rnk    = (float*)arena(R * sizeof(float));

    // setup
    k_setup_counts<<<1, 256, 0, stream>>>(mask, counts);
    k_setup_vecs<<<32, 256, 0, stream>>>(mask, counts, log_mu, u, v);

    // q/k/v linears (bf16 WMMA, f32 accumulate)
    dim3 gl(128, 5);
    k_linear_bf16<<<gl, 256, 0, stream>>>(xq, Wq, bq, qbf);
    k_linear_bf16<<<gl, 256, 0, stream>>>(xk, Wk, bk, kbf);
    k_linear_bf16<<<gl, 256, 0, stream>>>(xv, Wv, bv, vbf);
    k_rownorm<<<1024, 256, 0, stream>>>(qbf, rnq);
    k_rownorm<<<1024, 256, 0, stream>>>(kbf, rnk);

    // sim + simT (both L2-resident for the Sinkhorn passes)
    k_sim<<<dim3(16, 8, 8), 256, 0, stream>>>(kbf, qbf, rnk, rnq, sim, simT);

    // 100 Sinkhorn iterations (u then v each iteration)
    for (int it = 0; it < 100; ++it) {
        k_u_update<<<1024, 256, 0, stream>>>(sim, mask, u, v, log_mu);
        k_v_update<<<1024, 256, 0, stream>>>(simT, mask, u, v);
    }

    // x = T^T @ v  -> bf16 ; out = x @ Wp^T + bp ; attn_save
    k_tv<<<dim3(16, 5, 8), 256, 0, stream>>>(simT, vbf, mask, u, v, xbf);
    k_proj_f32<<<gl, 256, 0, stream>>>(xbf, Wp, bp, out);
    k_attn<<<1024, 256, 0, stream>>>(sim, mask, u, v, attn);
}
